// PatchDecoder_21577915695313
// MI455X (gfx1250) — matrix operand order repaired and checked by reading, not compiled, not run on hardware
//
#include <hip/hip_runtime.h>
#include <hip/hip_bf16.h>
#include <type_traits>
#include <cstdint>
#include <cstddef>

// Problem constants (match reference)
#define D_  512
#define H_  8
#define HD_ 64
#define PS_ 4
#define B_  4
#define T_  4096
#define P_  1024

#define USE_TDM 1

typedef __attribute__((ext_vector_type(16))) __bf16 bf16x16;
typedef __attribute__((ext_vector_type(8)))  __bf16 bf16x8;
typedef __attribute__((ext_vector_type(8)))  float  v8f;

typedef unsigned int u32x4 __attribute__((ext_vector_type(4)));
typedef int          i32x4 __attribute__((ext_vector_type(4)));
typedef int          i32x8 __attribute__((ext_vector_type(8)));

union BF16Frag {
    bf16x16 v;
    bf16x8  h[2];
    unsigned short u[16];
};

// fp32 -> bf16 with round-to-nearest-even
static __device__ __forceinline__ unsigned short f2bf(float f) {
    union { float f; unsigned u; } x; x.f = f;
    unsigned u = x.u;
    unsigned r = u + 0x7FFFu + ((u >> 16) & 1u);
    return (unsigned short)(r >> 16);
}

static __device__ __forceinline__ v8f v8f_zero() {
    v8f z;
    #pragma unroll
    for (int i = 0; i < 8; ++i) z[i] = 0.0f;
    return z;
}

static __device__ __forceinline__ v8f wmma_bf16(const BF16Frag& a, const BF16Frag& b, v8f c) {
    // D(f32 16x16) = A(bf16 16x32) * B(bf16 32x16) + C
    return __builtin_amdgcn_wmma_f32_16x16x32_bf16(false, a.v, false, b.v, (short)0, c, false, false);
}

#if USE_TDM
// ---------------------------------------------------------------------------
// Tensor Data Mover: async 2D tile load Global -> LDS (TENSORcnt-tracked).
// D# packing per CDNA5 ISA ch.8: group0 = {count, lds_addr, global_addr, type},
// group1 = {data_size, pad ctl, tensor dims, tile dims, strides}. data_size=2B.
// pad_interval_code: pad after 2^(c+1) DWORDs; pad_amount_code: (c+1) DWORDs.
// All arguments are block-uniform -> SGPR descriptors. Issued by one wave.
// ---------------------------------------------------------------------------
static __device__ __forceinline__ void tdm_load_2d_bf16(
    const void* gptr, unsigned lds_off,
    unsigned tensor_d0, unsigned tensor_d1, unsigned stride0_elems,
    unsigned tile_d0, unsigned tile_d1,
    unsigned pad_interval_code, unsigned pad_amount_code)
{
    const unsigned long long ga = (unsigned long long)(uintptr_t)gptr;
    u32x4 g0;
    g0[0] = 1u;                                        // [1:0] count=1 (valid D#)
    g0[1] = lds_off;                                   // [63:32] lds_addr
    g0[2] = (unsigned)ga;                              // [95:64] global_addr lo
    g0[3] = (unsigned)(ga >> 32) | (2u << 30);         // addr hi + [127:126] type=2
    i32x8 g1;
    g1[0] = (int)((1u << 16)                           // [17:16] data_size=1 (2B)
                | (1u << 20)                           // [20] pad_enable
                | (pad_interval_code << 22)            // [24:22]
                | (pad_amount_code  << 25));           // [31:25]
    g1[1] = (int)((tensor_d0 & 0xFFFFu) << 16);        // [79:48] tensor_dim0 (lo16)
    g1[2] = (int)((tensor_d0 >> 16) | ((tensor_d1 & 0xFFFFu) << 16)); // dim0 hi / dim1 lo
    g1[3] = (int)((tensor_d1 >> 16) | ((tile_d0 & 0xFFFFu) << 16));   // [127:112] tile_dim0
    g1[4] = (int)(tile_d1 & 0xFFFFu);                  // [143:128] tile_dim1, tile_dim2=0
    g1[5] = (int)stride0_elems;                        // [207:160] tensor_dim0_stride lo32
    g1[6] = 0;                                         // stride0 hi16 / dim1_stride lo16
    g1[7] = 0;
    const i32x4 z4 = {0, 0, 0, 0};
#if defined(__clang_major__) && (__clang_major__ >= 23)
    const i32x8 z8 = {0, 0, 0, 0, 0, 0, 0, 0};
    __builtin_amdgcn_tensor_load_to_lds(g0, g1, z4, z4, z8, 0);
#else
    __builtin_amdgcn_tensor_load_to_lds(g0, g1, z4, z4, 0);
#endif
}
#endif

// ---------------------------------------------------------------------------
// Generic 512-K GEMM: C[M,512] = A[M,512] * W[512,512] (+bias)(+residual)
// MODE 0: bf16 row-major output, scaled by oscale (Q gets 1/sqrt(HD) folded in)
// MODE 1: bf16 transposed per-head V output: Vt[b][h][d][p]
// MODE 2: fp32 output = acc + bias + residual (out projection)
// Block tile 128x128, 8 waves of 32; wave tile 32x64 (2x4 WMMA accumulators).
// ---------------------------------------------------------------------------
template<int MODE, typename AT>
__global__ __launch_bounds__(256)
void gemm512_kernel(const AT* __restrict__ A, const float* __restrict__ W,
                    const float* __restrict__ bias, const float* __restrict__ resid,
                    void* __restrict__ outp, float oscale)
{
    constexpr int BM = 128, BN = 128, BK = 32, LDK = BK + 8;
    __shared__ __attribute__((aligned(16))) unsigned short As[BM][LDK];  // A tile, k-major
    __shared__ __attribute__((aligned(16))) unsigned short Bs[BN][LDK];  // W^T tile: Bs[n][k]

    const int tid  = threadIdx.x;
    const int wave = tid >> 5;
    const int lane = tid & 31;
    const int l16  = lane & 15;
    const int lh   = lane >> 4;
    const int wm   = wave >> 1;              // 0..3  (M direction)
    const int wn   = wave & 1;               // 0..1  (N direction)
    const int bm0  = blockIdx.x * BM;
    const int bn0  = blockIdx.y * BN;

    v8f acc[2][4];
    #pragma unroll
    for (int mt = 0; mt < 2; ++mt)
        #pragma unroll
        for (int nt = 0; nt < 4; ++nt) acc[mt][nt] = v8f_zero();

    for (int kk = 0; kk < D_; kk += BK) {
        // --- stage A tile (128 x 32), fp32 (or bf16) global -> bf16 LDS ---
        #pragma unroll
        for (int c = 0; c < 4; ++c) {
            int idx  = tid + c * 256;
            int row  = idx >> 3;
            int col4 = (idx & 7) * 4;
            if constexpr (std::is_same<AT, float>::value) {
                const float4 v = *(const float4*)&A[(size_t)(bm0 + row) * D_ + kk + col4];
                ushort4 s;
                s.x = f2bf(v.x); s.y = f2bf(v.y); s.z = f2bf(v.z); s.w = f2bf(v.w);
                *(ushort4*)&As[row][col4] = s;
            } else {
                *(ushort4*)&As[row][col4] =
                    *(const ushort4*)&A[(size_t)(bm0 + row) * D_ + kk + col4];
            }
        }
        // --- stage W tile (32 x 128) transposed into Bs[n][k] ---
        #pragma unroll
        for (int c = 0; c < 4; ++c) {
            int idx  = tid + c * 256;
            int kr   = idx >> 5;
            int col4 = (idx & 31) * 4;
            const float4 v = *(const float4*)&W[(size_t)(kk + kr) * D_ + bn0 + col4];
            Bs[col4 + 0][kr] = f2bf(v.x);
            Bs[col4 + 1][kr] = f2bf(v.y);
            Bs[col4 + 2][kr] = f2bf(v.z);
            Bs[col4 + 3][kr] = f2bf(v.w);
        }
        __syncthreads();

        BF16Frag af[2], bf[4];
        #pragma unroll
        for (int mt = 0; mt < 2; ++mt) {
            int m  = wm * 32 + mt * 16 + l16;
            int ko = lh * 8;
            af[mt].h[0] = *(const bf16x8*)&As[m][ko];
            af[mt].h[1] = *(const bf16x8*)&As[m][ko + 16];
        }
        #pragma unroll
        for (int nt = 0; nt < 4; ++nt) {
            int n  = wn * 64 + nt * 16 + l16;
            int ko = lh * 8;
            bf[nt].h[0] = *(const bf16x8*)&Bs[n][ko];
            bf[nt].h[1] = *(const bf16x8*)&Bs[n][ko + 16];
        }
        #pragma unroll
        for (int mt = 0; mt < 2; ++mt)
            #pragma unroll
            for (int nt = 0; nt < 4; ++nt)
                acc[mt][nt] = wmma_bf16(af[mt], bf[nt], acc[mt][nt]);
        __syncthreads();
    }

    #pragma unroll
    for (int mt = 0; mt < 2; ++mt) {
        #pragma unroll
        for (int nt = 0; nt < 4; ++nt) {
            const int r0 = bm0 + wm * 32 + mt * 16 + lh * 8;
            const int c  = bn0 + wn * 64 + nt * 16 + l16;
            const float bc = bias[c];
            if constexpr (MODE == 0) {
                unsigned short* O = (unsigned short*)outp;
                #pragma unroll
                for (int j = 0; j < 8; ++j)
                    O[(size_t)(r0 + j) * D_ + c] = f2bf((acc[mt][nt][j] + bc) * oscale);
            } else if constexpr (MODE == 1) {
                unsigned short* O = (unsigned short*)outp;
                const int bidx = r0 >> 10;
                const int p0   = r0 & (P_ - 1);
                const int hh   = c >> 6;
                const int d    = c & (HD_ - 1);
                union { uint4 q; unsigned short s[8]; } u;
                #pragma unroll
                for (int j = 0; j < 8; ++j) u.s[j] = f2bf(acc[mt][nt][j] + bc);
                *(uint4*)&O[((size_t)((bidx * H_ + hh) * HD_ + d)) * P_ + p0] = u.q;
            } else {
                float* O = (float*)outp;
                #pragma unroll
                for (int j = 0; j < 8; ++j) {
                    const size_t off = (size_t)(r0 + j) * D_ + c;
                    O[off] = acc[mt][nt][j] + bc + resid[off];
                }
            }
        }
    }
}

// ---------------------------------------------------------------------------
// Flash attention, block-causal token->patch mask (token i sees p <= i/4).
// grid: (T/128, B*H). All 8 waves share (b,h) => K/V tiles staged ONCE per
// block into LDS via TDM (double-buffered, TENSORcnt). Q is pre-scaled by
// 1/sqrt(HD). Only the last tile is a mask boundary for any wave in a block.
// Row sums computed by an extra WMMA against an all-ones B matrix (no
// cross-lane shuffles); only the row max uses a 4-step butterfly.
// ---------------------------------------------------------------------------
__global__ __launch_bounds__(256)
void attn_kernel(const unsigned short* __restrict__ Q,
                 const unsigned short* __restrict__ K,
                 const unsigned short* __restrict__ Vt,
                 unsigned short* __restrict__ Ctx)
{
    // TDM pads 4 DWORDs (8 bf16) after each row -> row strides 72 / 40.
    __shared__ __attribute__((aligned(16))) unsigned short Ks[2][32][HD_ + 8]; // [buf][patch][d]
    __shared__ __attribute__((aligned(16))) unsigned short Vs[2][HD_][32 + 8]; // [buf][d][patch]
    __shared__ __attribute__((aligned(16))) unsigned short Psm[8][16][40];

    const int tid  = threadIdx.x;
    const int wave = tid >> 5;
    const int lane = tid & 31;
    const int l16  = lane & 15;
    const int lh   = lane >> 4;
    const int bh   = blockIdx.y;
    const int b    = bh >> 3;
    const int h    = bh & 7;
    const int tbase  = blockIdx.x * 128 + wave * 16;
    const int ntiles = blockIdx.x + 1;           // block-uniform allowed tile count

    // Q fragments for this wave's 16 rows (HD=64 -> two 16x32 A fragments)
    BF16Frag qf[2];
    #pragma unroll
    for (int kc = 0; kc < 2; ++kc) {
        const size_t base = (size_t)(b * T_ + tbase + l16) * D_ + h * HD_ + kc * 32;
        qf[kc].h[0] = *(const bf16x8*)&Q[base + lh * 8];
        qf[kc].h[1] = *(const bf16x8*)&Q[base + lh * 8 + 16];
    }

    // Constant all-ones B fragment: rowsum(P) = P @ ones via one WMMA
    BF16Frag ones;
    #pragma unroll
    for (int i = 0; i < 16; ++i) ones.u[i] = 0x3F80u;   // 1.0bf16

    v8f acc[4];
    #pragma unroll
    for (int nt = 0; nt < 4; ++nt) acc[nt] = v8f_zero();
    float mrow[8], lrow[8];
    #pragma unroll
    for (int j = 0; j < 8; ++j) { mrow[j] = -3.0e38f; lrow[j] = 0.0f; }

#if USE_TDM
    // K tile: 32 patches x 64 dims, row = 32 DWORDs -> interval code 4, pad 4 DW -> code 3
    // V tile: 64 dims x 32 patches, row = 16 DWORDs -> interval code 3, pad 4 DW -> code 3
    auto issue_tiles = [&](int pt, int buf) {
        const int pbase = pt * 32;
        tdm_load_2d_bf16(&K [(size_t)(b * P_ + pbase) * D_ + h * HD_],
                         (unsigned)(uintptr_t)&Ks[buf][0][0],
                         HD_, 32, D_,  HD_, 32, 4, 3);
        tdm_load_2d_bf16(&Vt[((size_t)((b * H_ + h) * HD_)) * P_ + pbase],
                         (unsigned)(uintptr_t)&Vs[buf][0][0],
                         32, HD_, P_,  32, HD_, 3, 3);
    };
    if (wave == 0) issue_tiles(0, 0);
#else
    auto stage_tiles = [&](int pt, int buf) {
        const int pbase = pt * 32;
        {   // K: 32x64 bf16, one uint4 (8 bf16) per thread
            const int row = tid >> 3, c8 = (tid & 7) * 8;
            *(uint4*)&Ks[buf][row][c8] =
                *(const uint4*)&K[(size_t)(b * P_ + pbase + row) * D_ + h * HD_ + c8];
        }
        {   // Vt: 64x32 bf16
            const int row = tid >> 2, c8 = (tid & 3) * 8;
            *(uint4*)&Vs[buf][row][c8] =
                *(const uint4*)&Vt[((size_t)((b * H_ + h) * HD_ + row)) * P_ + pbase + c8];
        }
    };
    stage_tiles(0, 0);
#endif

    for (int pt = 0; pt < ntiles; ++pt) {
        const int buf = pt & 1;
#if USE_TDM
        if (pt + 1 < ntiles) {
            if (wave == 0) {
                issue_tiles(pt + 1, (pt + 1) & 1);
                __builtin_amdgcn_s_wait_tensorcnt(2);   // current 2 tiles done
            }
        } else if (wave == 0) {
            __builtin_amdgcn_s_wait_tensorcnt(0);
        }
        __syncthreads();
#else
        if (pt + 1 < ntiles) stage_tiles(pt + 1, (pt + 1) & 1);
        __syncthreads();
#endif

        const int pbase  = pt * 32;
        const bool boundary = (pt == ntiles - 1);   // only tile needing the mask

        // K^T fragments from LDS: col = patch, k = head dim (contiguous)
        BF16Frag kf[2][2];
        #pragma unroll
        for (int st = 0; st < 2; ++st) {
            const int patch = st * 16 + l16;
            #pragma unroll
            for (int kc = 0; kc < 2; ++kc) {
                kf[st][kc].h[0] = *(const bf16x8*)&Ks[buf][patch][kc * 32 + lh * 8];
                kf[st][kc].h[1] = *(const bf16x8*)&Ks[buf][patch][kc * 32 + lh * 8 + 16];
            }
        }
        // V fragments issued early: ds loads overlap the softmax VALU section
        BF16Frag vf[4];
        #pragma unroll
        for (int nt = 0; nt < 4; ++nt) {
            const int d = nt * 16 + l16;
            vf[nt].h[0] = *(const bf16x8*)&Vs[buf][d][lh * 8];
            vf[nt].h[1] = *(const bf16x8*)&Vs[buf][d][lh * 8 + 16];
        }

        v8f s0 = wmma_bf16(qf[0], kf[0][0], v8f_zero());
        s0     = wmma_bf16(qf[1], kf[0][1], s0);
        v8f s1 = wmma_bf16(qf[0], kf[1][0], v8f_zero());
        s1     = wmma_bf16(qf[1], kf[1][1], s1);

        // online softmax; D-layout: element j -> row tbase+lh*8+j, lane -> col l16
        #pragma unroll
        for (int j = 0; j < 8; ++j) {
            float v0 = s0[j];
            float v1 = s1[j];
            if (boundary) {
                const int pa = (tbase + lh * 8 + j) >> 2;
                if (pbase + l16 > pa)      v0 = -3.0e38f;
                if (pbase + 16 + l16 > pa) v1 = -3.0e38f;
            }

            float rm = fmaxf(v0, v1);
            #pragma unroll
            for (int msk = 1; msk < 16; msk <<= 1)
                rm = fmaxf(rm, __shfl_xor(rm, msk, 32));
            const float mn    = fmaxf(mrow[j], rm);
            const float alpha = __expf(mrow[j] - mn);
            mrow[j] = mn;
            const float p0 = __expf(v0 - mn);
            const float p1 = __expf(v1 - mn);
            lrow[j] *= alpha;                       // row sum added via WMMA below
            #pragma unroll
            for (int nt = 0; nt < 4; ++nt) acc[nt][j] *= alpha;

            const int rr = lh * 8 + j;
            Psm[wave][rr][l16]      = f2bf(p0);
            Psm[wave][rr][16 + l16] = f2bf(p1);
        }
        asm volatile("s_wait_dscnt 0" ::: "memory");  // wave-local LDS RAW fence

        // re-read P in A-fragment layout
        BF16Frag pf;
        pf.h[0] = *(const bf16x8*)&Psm[wave][l16][lh * 8];
        pf.h[1] = *(const bf16x8*)&Psm[wave][l16][lh * 8 + 16];

        // row sums: P(16x32) @ ones(32x16) -> every column of row r = sum_k P[r,k]
        const v8f rs8 = wmma_bf16(pf, ones, v8f_zero());
        #pragma unroll
        for (int j = 0; j < 8; ++j) lrow[j] += rs8[j];

        // ctx += P(16x32) @ V(32x64)
        #pragma unroll
        for (int nt = 0; nt < 4; ++nt)
            acc[nt] = wmma_bf16(pf, vf[nt], acc[nt]);
        __syncthreads();   // all waves done with buf before it is refilled
    }

    // normalize and store ctx (bf16) to [B,T,D] at head slice
    #pragma unroll
    for (int j = 0; j < 8; ++j) {
        const float inv = 1.0f / lrow[j];
        const int tok = tbase + lh * 8 + j;
        #pragma unroll
        for (int nt = 0; nt < 4; ++nt)
            Ctx[(size_t)(b * T_ + tok) * D_ + h * HD_ + nt * 16 + l16] =
                f2bf(acc[nt][j] * inv);
    }
}

// ---------------------------------------------------------------------------
extern "C" void kernel_launch(void* const* d_in, const int* in_sizes, int n_in,
                              void* d_out, int out_size, void* d_ws, size_t ws_size,
                              hipStream_t stream)
{
    (void)in_sizes; (void)n_in; (void)out_size; (void)ws_size;

    const float* tok = (const float*)d_in[0];
    const float* pat = (const float*)d_in[1];
    const float* Wq  = (const float*)d_in[2];
    const float* Wk  = (const float*)d_in[3];
    const float* Wv  = (const float*)d_in[4];
    const float* bq  = (const float*)d_in[5];
    const float* bk  = (const float*)d_in[6];
    const float* bv  = (const float*)d_in[7];
    const float* Wo  = (const float*)d_in[8];
    const float* bo  = (const float*)d_in[9];

    char* ws = (char*)d_ws;
    unsigned short* Qbf = (unsigned short*)(ws);                              // 16 MB
    unsigned short* Kbf = (unsigned short*)(ws + (size_t)16 * 1024 * 1024);   //  4 MB
    unsigned short* Vt  = (unsigned short*)(ws + (size_t)20 * 1024 * 1024);   //  4 MB
    unsigned short* Ctx = (unsigned short*)(ws + (size_t)24 * 1024 * 1024);   // 16 MB

    dim3 blk(256);
    // Q = (tok @ Wq + bq) * (1/sqrt(HD))  -> bf16 [B*T, D]   (scale folded in)
    gemm512_kernel<0, float><<<dim3(128, 4), blk, 0, stream>>>(tok, Wq, bq, nullptr, Qbf, 0.125f);
    // K = pat @ Wk + bk  -> bf16 [B*P, D]
    gemm512_kernel<0, float><<<dim3(32, 4),  blk, 0, stream>>>(pat, Wk, bk, nullptr, Kbf, 1.0f);
    // V = pat @ Wv + bv  -> bf16 transposed Vt[b][h][d][p]
    gemm512_kernel<1, float><<<dim3(32, 4),  blk, 0, stream>>>(pat, Wv, bv, nullptr, Vt,  1.0f);
    // flash cross-attention with block-causal mask
    attn_kernel<<<dim3(32, 32), blk, 0, stream>>>(Qbf, Kbf, Vt, Ctx);
    // out = ctx @ Wo + bo + tok  -> fp32 [B,T,D]
    gemm512_kernel<2, unsigned short><<<dim3(128, 4), blk, 0, stream>>>(Ctx, Wo, bo, tok, d_out, 1.0f);
}
